// BartDecoderLayer_27779848471448
// MI455X (gfx1250) — matrix operand order repaired and checked by reading, not compiled, not run on hardware
//
#include <hip/hip_runtime.h>
#include <hip/hip_bf16.h>
#include <math.h>

// ---------------------------------------------------------------------------
// Types for CDNA5 WMMA (wave32): D(16x16,f32) = A(16x32,bf16) x B(32x16,bf16) + C
// ---------------------------------------------------------------------------
typedef __attribute__((ext_vector_type(8)))  __bf16 v8bf;
typedef __attribute__((ext_vector_type(16))) __bf16 v16bf;
typedef __attribute__((ext_vector_type(8)))  float  v8f;

#if defined(__HIP_DEVICE_COMPILE__) && __has_builtin(__builtin_amdgcn_global_load_async_to_lds_b128)
#define HAVE_ASYNC_LDS 1
// exact pointee type expected by the builtin (from hipcc diagnostic):
//   __attribute__((__vector_size__(4*sizeof(int)))) int  in AS(1)/AS(3)
typedef int v4i_gcc __attribute__((__vector_size__(16)));
typedef __attribute__((address_space(1))) v4i_gcc* gas_v4i_ptr;
typedef __attribute__((address_space(3))) v4i_gcc* las_v4i_ptr;
#endif

__device__ __forceinline__ v16bf make16(v8bf lo, v8bf hi) {
    union { v16bf v; v8bf h[2]; } u;
    u.h[0] = lo; u.h[1] = hi;
    return u.v;
}

__device__ __forceinline__ v8f zero8() {
    v8f z;
#pragma unroll
    for (int i = 0; i < 8; ++i) z[i] = 0.0f;
    return z;
}

__device__ __forceinline__ v8f wmma_bf16(v16bf a, v16bf b, v8f c) {
    // emits v_wmma_f32_16x16x32_bf16
    return __builtin_amdgcn_wmma_f32_16x16x32_bf16(
        /*neg_a=*/false, a, /*neg_b=*/false, b,
        /*c_mod=*/(short)0, c, /*reuse_a=*/false, /*reuse_b=*/false);
}

__device__ __forceinline__ void wait_async0() {
#if defined(HAVE_ASYNC_LDS)
#if __has_builtin(__builtin_amdgcn_s_wait_asynccnt)
    __builtin_amdgcn_s_wait_asynccnt(0);
#else
    asm volatile("s_wait_asynccnt 0" ::: "memory");
#endif
#endif
}

// ---------------------------------------------------------------------------
// fp32 -> bf16 conversion (grid covers n/4 threads; all our n are %4==0)
// ---------------------------------------------------------------------------
__global__ __launch_bounds__(256) void cvt_kernel(const float* __restrict__ in,
                                                  __bf16* __restrict__ out, int n) {
    int i = (blockIdx.x * 256 + threadIdx.x) * 4;
    if (i + 3 < n) {
        float4 v = *(const float4*)(in + i);
        out[i + 0] = (__bf16)v.x;
        out[i + 1] = (__bf16)v.y;
        out[i + 2] = (__bf16)v.z;
        out[i + 3] = (__bf16)v.w;
    } else {
        for (int j = i; j < n; ++j) out[j] = (__bf16)in[j];
    }
}

// ---------------------------------------------------------------------------
// GEMM: C[M,N] = A[M,K] * W[N,K]^T (+bias, *scale, optional exact GELU)
// A, W bf16; accum f32. Outputs: optional f32 [M,N], optional bf16 [M,N]
// or bf16 transposed-V layout [B*H, 64, vtT] (vt != 0).
// Block: 256 thr = 8 waves, tile 128x128, BK=32. Wave patch 32(M) x 64(N).
// Double-buffered LDS, one barrier per K-step; staging via CDNA5
// global_load_async_to_lds_b128 when available (ASYNCcnt pipeline).
// ---------------------------------------------------------------------------
__global__ __launch_bounds__(256) void gemm_bf16_kernel(
    const __bf16* __restrict__ A, const __bf16* __restrict__ W,
    const float* __restrict__ bias,
    float* __restrict__ outF, __bf16* __restrict__ outB,
    int M, int N, int K, float scale, int dogelu, int vt, int vtT) {
    __shared__ __bf16 sA[2][128][40];   // 32 + 8 pad per row (16B-aligned, no conflicts)
    __shared__ __bf16 sW[2][128][40];

    const int tid  = threadIdx.x;
    const int wave = tid >> 5;
    const int lane = tid & 31;
    const int wm = (wave >> 1) * 32;   // 4 waves along M
    const int wn = (wave & 1) * 64;    // 2 waves along N
    const int m0 = blockIdx.y * 128;
    const int n0 = blockIdx.x * 128;
    const int lrow = lane & 15;
    const int klo  = (lane < 16) ? 0 : 8;    // A-frag K split (ISA 7.12.2)
    const int kb   = (lane < 16) ? 0 : 8;    // B-frag K split
    const int rofs = (lane < 16) ? 0 : 8;    // C/D row split

    v8f acc[2][4];
#pragma unroll
    for (int mt = 0; mt < 2; ++mt)
#pragma unroll
        for (int nt = 0; nt < 4; ++nt) acc[mt][nt] = zero8();

    const int nk = K >> 5;

    auto compute = [&](int buf) {
        v16bf afr[2], bfr[4];
#pragma unroll
        for (int mt = 0; mt < 2; ++mt) {
            const __bf16* p = &sA[buf][wm + mt * 16 + lrow][0];
            afr[mt] = make16(*(const v8bf*)&p[klo], *(const v8bf*)&p[16 + klo]);
        }
#pragma unroll
        for (int nt = 0; nt < 4; ++nt) {
            const __bf16* p = &sW[buf][wn + nt * 16 + lrow][0];
            bfr[nt] = make16(*(const v8bf*)&p[kb], *(const v8bf*)&p[kb + 16]);
        }
#pragma unroll
        for (int mt = 0; mt < 2; ++mt)
#pragma unroll
            for (int nt = 0; nt < 4; ++nt)
                acc[mt][nt] = wmma_bf16(afr[mt], bfr[nt], acc[mt][nt]);
    };

#if defined(HAVE_ASYNC_LDS)
    // ---- async global->LDS pipeline (no VGPR round-trip) ----
    auto stage_async = [&](int buf, int k0) {
#pragma unroll
        for (int t = 0; t < 2; ++t) {
            int vec = tid + t * 256;          // 0..511
            int r   = vec >> 2;               // 0..127
            int c8  = (vec & 3) * 8;          // 0,8,16,24
            __builtin_amdgcn_global_load_async_to_lds_b128(
                (gas_v4i_ptr)&A[(size_t)(m0 + r) * K + k0 + c8],
                (las_v4i_ptr)&sA[buf][r][c8], 0, 0);
            __builtin_amdgcn_global_load_async_to_lds_b128(
                (gas_v4i_ptr)&W[(size_t)(n0 + r) * K + k0 + c8],
                (las_v4i_ptr)&sW[buf][r][c8], 0, 0);
        }
    };
    stage_async(0, 0);
    for (int kt = 0; kt < nk; ++kt) {
        const int cur = kt & 1;
        wait_async0();                 // my tile-kt async loads landed in LDS
        __syncthreads();               // everyone's tile-kt visible; prior reads done
        if (kt + 1 < nk) stage_async(cur ^ 1, (kt + 1) << 5);
        compute(cur);
    }
#else
    // ---- fallback: register-staged double buffer, one barrier per step ----
    v8bf rA[2], rW[2];
    auto gload = [&](int k0) {
#pragma unroll
        for (int t = 0; t < 2; ++t) {
            int vec = tid + t * 256;
            int r   = vec >> 2;
            int c8  = (vec & 3) * 8;
            rA[t] = *(const v8bf*)&A[(size_t)(m0 + r) * K + k0 + c8];
            rW[t] = *(const v8bf*)&W[(size_t)(n0 + r) * K + k0 + c8];
        }
    };
    auto push = [&](int buf) {
#pragma unroll
        for (int t = 0; t < 2; ++t) {
            int vec = tid + t * 256;
            int r   = vec >> 2;
            int c8  = (vec & 3) * 8;
            *(v8bf*)&sA[buf][r][c8] = rA[t];
            *(v8bf*)&sW[buf][r][c8] = rW[t];
        }
    };
    gload(0);
    for (int kt = 0; kt < nk; ++kt) {
        const int cur = kt & 1;
        push(cur);
        if (kt + 1 < nk) gload((kt + 1) << 5);   // overlap next tile with compute
        __syncthreads();
        compute(cur);
    }
#endif

    // epilogue: bias -> (gelu) -> scale -> store(s)
#pragma unroll
    for (int mt = 0; mt < 2; ++mt) {
#pragma unroll
        for (int nt = 0; nt < 4; ++nt) {
            const int col = n0 + wn + nt * 16 + lrow;
            const float bv = bias[col];
#pragma unroll
            for (int r = 0; r < 8; ++r) {
                const int row = m0 + wm + mt * 16 + r + rofs;
                float val = acc[mt][nt][r] + bv;
                if (dogelu) val = 0.5f * val * (1.0f + erff(val * 0.70710678118654752f));
                val *= scale;
                if (outF) outF[(size_t)row * N + col] = val;
                if (outB) {
                    if (vt) {
                        // transposed-V store: [B*H, 64, vtT]; H=16, HD=64
                        int bidx = row / vtT;
                        int t    = row - bidx * vtT;
                        int h    = col >> 6;
                        int hd   = col & 63;
                        outB[(((size_t)bidx * 16 + h) * 64 + hd) * (size_t)vtT + t] = (__bf16)val;
                    } else {
                        outB[(size_t)row * N + col] = (__bf16)val;
                    }
                }
            }
        }
    }
}

// ---------------------------------------------------------------------------
// Flash attention (online softmax), D=1024, H=16, HD=64, wave32 WMMA.
// Q [B,Tq,D] bf16 (pre-scaled), K [B,Tk,D] bf16, VT [B*H,64,Tk] bf16,
// mask [B,1,Tq,Tk] f32 additive, O [B,Tq,D] bf16.
// Block: 128 thr = 4 waves; wave owns 16 query rows; 32 keys per iteration.
// ---------------------------------------------------------------------------
__global__ __launch_bounds__(128) void attn_kernel(
    const __bf16* __restrict__ Q, const __bf16* __restrict__ Kv,
    const __bf16* __restrict__ VT, const float* __restrict__ mask,
    __bf16* __restrict__ O, int Tq, int Tk) {
    __shared__ __bf16 sP[4][16][40];  // per-wave P tile (16 x 32, padded)

    const int lane = threadIdx.x & 31;
    const int wave = threadIdx.x >> 5;
    const int bh = blockIdx.y;
    const int b  = bh >> 4;      // H = 16
    const int h  = bh & 15;
    const int qbase = blockIdx.x * 64 + wave * 16;
    const int lrow = lane & 15;
    const int klo  = (lane < 16) ? 0 : 8;
    const int kb   = (lane < 16) ? 0 : 8;
    const int rofs = (lane < 16) ? 0 : 8;

    // Q fragments (16 rows x 64 hd) stay in registers for the whole pass
    v16bf aq[2];
    {
        const __bf16* qp = Q + ((size_t)(b * Tq + qbase + lrow)) * 1024 + h * 64;
#pragma unroll
        for (int c = 0; c < 2; ++c)
            aq[c] = make16(*(const v8bf*)&qp[c * 32 + klo],
                           *(const v8bf*)&qp[c * 32 + 16 + klo]);
    }

    v8f accO[4];
#pragma unroll
    for (int c = 0; c < 4; ++c) accO[c] = zero8();
    float mrow[8], lsum[8];
#pragma unroll
    for (int r = 0; r < 8; ++r) { mrow[r] = -1e30f; lsum[r] = 0.0f; }

    const float* mp = mask + (size_t)b * Tq * Tk;

    for (int kt = 0; kt < Tk; kt += 32) {
        // ---- scores S = Q * K^T (+mask) for two 16-key groups ----
        v8f sfr[2];
#pragma unroll
        for (int g = 0; g < 2; ++g) {
            sfr[g] = zero8();
            const __bf16* kp = Kv + ((size_t)(b * Tk + kt + g * 16 + lrow)) * 1024 + h * 64;
#pragma unroll
            for (int c = 0; c < 2; ++c) {
                v16bf bk = make16(*(const v8bf*)&kp[c * 32 + kb],
                                  *(const v8bf*)&kp[c * 32 + kb + 16]);
                sfr[g] = wmma_bf16(aq[c], bk, sfr[g]);
            }
#pragma unroll
            for (int r = 0; r < 8; ++r)
                sfr[g][r] += mp[(size_t)(qbase + r + rofs) * Tk + kt + g * 16 + lrow];
        }

        // ---- online softmax: cross-lane max/sum within 16-lane row groups ----
        float lm[8];
#pragma unroll
        for (int r = 0; r < 8; ++r) lm[r] = fmaxf(sfr[0][r], sfr[1][r]);
#pragma unroll
        for (int d = 1; d < 16; d <<= 1)
#pragma unroll
            for (int r = 0; r < 8; ++r) lm[r] = fmaxf(lm[r], __shfl_xor(lm[r], d));

        float al[8], p0[8], p1[8], ls[8];
#pragma unroll
        for (int r = 0; r < 8; ++r) {
            float mn = fmaxf(mrow[r], lm[r]);
            al[r]   = __expf(mrow[r] - mn);
            mrow[r] = mn;
            p0[r] = __expf(sfr[0][r] - mn);
            p1[r] = __expf(sfr[1][r] - mn);
            ls[r] = p0[r] + p1[r];
        }
#pragma unroll
        for (int d = 1; d < 16; d <<= 1)
#pragma unroll
            for (int r = 0; r < 8; ++r) ls[r] += __shfl_xor(ls[r], d);
#pragma unroll
        for (int r = 0; r < 8; ++r) lsum[r] = lsum[r] * al[r] + ls[r];
#pragma unroll
        for (int c = 0; c < 4; ++c)
#pragma unroll
            for (int r = 0; r < 8; ++r) accO[c][r] *= al[r];

        // ---- C-layout -> A-layout for P via per-wave LDS tile ----
#pragma unroll
        for (int r = 0; r < 8; ++r) {
            sP[wave][r + rofs][lrow]      = (__bf16)p0[r];
            sP[wave][r + rofs][16 + lrow] = (__bf16)p1[r];
        }
        asm volatile("s_wait_dscnt 0" ::: "memory");
        v16bf ap = make16(*(const v8bf*)&sP[wave][lrow][klo],
                          *(const v8bf*)&sP[wave][lrow][16 + klo]);

        // ---- O += P * V (V transposed: contiguous B-fragments) ----
#pragma unroll
        for (int c = 0; c < 4; ++c) {
            const __bf16* vp = VT + ((size_t)bh * 64 + c * 16 + lrow) * (size_t)Tk + kt + kb;
            v16bf bv = make16(*(const v8bf*)&vp[0], *(const v8bf*)&vp[16]);
            accO[c] = wmma_bf16(ap, bv, accO[c]);
        }
    }

    // ---- normalize and store O ----
    float inv[8];
#pragma unroll
    for (int r = 0; r < 8; ++r) inv[r] = 1.0f / lsum[r];
#pragma unroll
    for (int c = 0; c < 4; ++c)
#pragma unroll
        for (int r = 0; r < 8; ++r)
            O[((size_t)(b * Tq + qbase + r + rofs)) * 1024 + h * 64 + c * 16 + lrow] =
                (__bf16)(accO[c][r] * inv[r]);
}

// ---------------------------------------------------------------------------
// y = LayerNorm(x + res) * g + b ; writes f32 (and optional bf16 copy)
// one block (256 thr) per row of D=1024
// ---------------------------------------------------------------------------
__global__ __launch_bounds__(256) void add_ln_kernel(
    const float* __restrict__ x, const float* __restrict__ res,
    const float* __restrict__ g, const float* __restrict__ bta,
    float* __restrict__ outF, __bf16* __restrict__ outB) {
    const int row = blockIdx.x;
    const int t = threadIdx.x;
    const float* xr = x + (size_t)row * 1024;
    const float* rr = res + (size_t)row * 1024;
    __shared__ float red[256];

    float v[4];
    float s = 0.0f;
#pragma unroll
    for (int i = 0; i < 4; ++i) { v[i] = xr[t + i * 256] + rr[t + i * 256]; s += v[i]; }
    red[t] = s; __syncthreads();
    for (int o = 128; o > 0; o >>= 1) { if (t < o) red[t] += red[t + o]; __syncthreads(); }
    float mean = red[0] * (1.0f / 1024.0f);
    __syncthreads();
    float s2 = 0.0f;
#pragma unroll
    for (int i = 0; i < 4; ++i) { float d = v[i] - mean; s2 += d * d; }
    red[t] = s2; __syncthreads();
    for (int o = 128; o > 0; o >>= 1) { if (t < o) red[t] += red[t + o]; __syncthreads(); }
    float rstd = rsqrtf(red[0] * (1.0f / 1024.0f) + 1e-5f);
#pragma unroll
    for (int i = 0; i < 4; ++i) {
        int col = t + i * 256;
        float y = (v[i] - mean) * rstd * g[col] + bta[col];
        outF[(size_t)row * 1024 + col] = y;
        if (outB) outB[(size_t)row * 1024 + col] = (__bf16)y;
    }
}

// ---------------------------------------------------------------------------
// Host orchestration
// ---------------------------------------------------------------------------
extern "C" void kernel_launch(void* const* d_in, const int* in_sizes, int n_in,
                              void* d_out, int out_size, void* d_ws, size_t ws_size,
                              hipStream_t stream) {
    (void)in_sizes; (void)n_in; (void)out_size; (void)ws_size;
    const int B_ = 4, T_ = 1024, S_ = 1024, D_ = 1024, H_ = 16, F_ = 4096;
    (void)H_;
    const size_t NBT = (size_t)B_ * T_ * D_;   // 4M activations

    const float* hidden = (const float*)d_in[0];
    const float* enc    = (const float*)d_in[1];
    const float* amask  = (const float*)d_in[2];
    const float* emask  = (const float*)d_in[3];
    const float* sa_wq = (const float*)d_in[4];  const float* sa_bq = (const float*)d_in[5];
    const float* sa_wk = (const float*)d_in[6];  const float* sa_bk = (const float*)d_in[7];
    const float* sa_wv = (const float*)d_in[8];  const float* sa_bv = (const float*)d_in[9];
    const float* sa_wo = (const float*)d_in[10]; const float* sa_bo = (const float*)d_in[11];
    const float* ca_wq = (const float*)d_in[12]; const float* ca_bq = (const float*)d_in[13];
    const float* ca_wk = (const float*)d_in[14]; const float* ca_bk = (const float*)d_in[15];
    const float* ca_wv = (const float*)d_in[16]; const float* ca_bv = (const float*)d_in[17];
    const float* ca_wo = (const float*)d_in[18]; const float* ca_bo = (const float*)d_in[19];
    const float* fc1_w = (const float*)d_in[20]; const float* fc1_b = (const float*)d_in[21];
    const float* fc2_w = (const float*)d_in[22]; const float* fc2_b = (const float*)d_in[23];
    const float* ln1_g = (const float*)d_in[24]; const float* ln1_b = (const float*)d_in[25];
    const float* ln2_g = (const float*)d_in[26]; const float* ln2_b = (const float*)d_in[27];
    const float* ln3_g = (const float*)d_in[28]; const float* ln3_b = (const float*)d_in[29];

    char* ws = (char*)d_ws;
    size_t off = 0;
    auto alloc = [&](size_t bytes) {
        size_t o = off;
        off = (off + bytes + 255) & ~(size_t)255;
        return o;
    };
    __bf16* xbf   = (__bf16*)(ws + alloc(NBT * 2));
    __bf16* encbf = (__bf16*)(ws + alloc(NBT * 2));
    __bf16* wbuf  = (__bf16*)(ws + alloc((size_t)F_ * D_ * 2));   // one weight staged at a time
    __bf16* qbf   = (__bf16*)(ws + alloc(NBT * 2));
    __bf16* kbf   = (__bf16*)(ws + alloc(NBT * 2));
    __bf16* vTb   = (__bf16*)(ws + alloc(NBT * 2));
    __bf16* abf   = (__bf16*)(ws + alloc(NBT * 2));
    float*  tmpf  = (float*)(ws + alloc(NBT * 4));
    float*  h1f   = (float*)(ws + alloc(NBT * 4));
    __bf16* h1bf  = (__bf16*)(ws + alloc(NBT * 2));
    float*  h2f   = (float*)(ws + alloc(NBT * 4));
    __bf16* h2bf  = (__bf16*)(ws + alloc(NBT * 2));
    __bf16* f1bf  = qbf;  // reuse the contiguous 32MB q/k/vT/a region for fc1 output

    auto cvt = [&](const float* src, __bf16* dst, size_t n) {
        int blocks = (int)((n / 4 + 255) / 256);
        cvt_kernel<<<blocks, 256, 0, stream>>>(src, dst, (int)n);
    };
    auto gemm = [&](const __bf16* A, const float* Wsrc, const float* bias,
                    int M, int N, int K, float* oF, __bf16* oB,
                    float scale, int gelu, int vt) {
        cvt(Wsrc, wbuf, (size_t)N * K);
        dim3 grid(N / 128, M / 128);
        gemm_bf16_kernel<<<grid, 256, 0, stream>>>(A, wbuf, bias, oF, oB,
                                                   M, N, K, scale, gelu, vt, T_);
    };

    const int M = B_ * T_;          // 4096
    const float qscale = 0.125f;    // HD^-0.5, HD=64

    cvt(hidden, xbf, NBT);
    cvt(enc, encbf, NBT);

    // ---- self attention ----
    gemm(xbf, sa_wq, sa_bq, M, D_, D_, nullptr, qbf, qscale, 0, 0);
    gemm(xbf, sa_wk, sa_bk, M, D_, D_, nullptr, kbf, 1.0f, 0, 0);
    gemm(xbf, sa_wv, sa_bv, M, D_, D_, nullptr, vTb, 1.0f, 0, 1);
    attn_kernel<<<dim3(T_ / 64, B_ * 16), 128, 0, stream>>>(qbf, kbf, vTb, amask, abf, T_, T_);
    gemm(abf, sa_wo, sa_bo, M, D_, D_, tmpf, nullptr, 1.0f, 0, 0);
    add_ln_kernel<<<M, 256, 0, stream>>>(tmpf, hidden, ln1_g, ln1_b, h1f, h1bf);

    // ---- cross attention ----
    gemm(h1bf, ca_wq, ca_bq, M, D_, D_, nullptr, qbf, qscale, 0, 0);
    gemm(encbf, ca_wk, ca_bk, M, D_, D_, nullptr, kbf, 1.0f, 0, 0);
    gemm(encbf, ca_wv, ca_bv, M, D_, D_, nullptr, vTb, 1.0f, 0, 1);
    attn_kernel<<<dim3(T_ / 64, B_ * 16), 128, 0, stream>>>(qbf, kbf, vTb, emask, abf, T_, S_);
    gemm(abf, ca_wo, ca_bo, M, D_, D_, tmpf, nullptr, 1.0f, 0, 0);
    add_ln_kernel<<<M, 256, 0, stream>>>(tmpf, h1f, ln2_g, ln2_b, h2f, h2bf);

    // ---- FFN ----
    gemm(h2bf, fc1_w, fc1_b, M, F_, D_, nullptr, f1bf, 1.0f, /*gelu=*/1, 0);
    gemm(f1bf, fc2_w, fc2_b, M, D_, F_, tmpf, nullptr, 1.0f, 0, 0);
    add_ln_kernel<<<M, 256, 0, stream>>>(tmpf, h2f, ln3_g, ln3_b, (float*)d_out, nullptr);
}
